// AttentionBlock_31679678775736
// MI455X (gfx1250) — compile-verified, matrix operand order checked by reading, not run on hardware
//
#include <hip/hip_runtime.h>
#include <hip/hip_bf16.h>

// ---------------------------------------------------------------------------
// CDNA5 (gfx1250) attention block, bf16 WMMA + async-to-LDS tile staging:
//   pass 0: x, Wqkv, Wout -> bf16 (one-time conversion)
//   pass 1: qkv = x @ Wqkv^T  (LDS double-buffered WMMA GEMM via
//           global_load_async_to_lds_b128); scatter q,k (B,H,S,D), vT (B,H,D,S)
//   pass 2: windowed flash attention (wl=512, wr=0) -> attn_out bf16
//   pass 3: out = attn_out @ Wout^T (same async-LDS WMMA GEMM) -> f32
// B=4, S=2048, E=1024, H=16, D=64
// ---------------------------------------------------------------------------

typedef __attribute__((ext_vector_type(16))) __bf16 bf16x16;
typedef __attribute__((ext_vector_type(8)))  __bf16 bf16x8;
typedef __attribute__((ext_vector_type(8)))  float  f32x8;

#define WB 4
#define WS 2048
#define WE 1024
#define WH 16
#define WD 64
#define WN3 3072

static __device__ __forceinline__ f32x8
wmma_bf16(bf16x16 a, bf16x16 b, f32x8 c) {
    return __builtin_amdgcn_wmma_f32_16x16x32_bf16(
        /*neg_a=*/false, a, /*neg_b=*/false, b,
        /*c_mod=*/(short)0, c, /*reuse_a=*/false, /*reuse_b=*/false);
}

// 16-byte async DMA: global -> LDS, tracked by ASYNCcnt (CDNA5 §10/§15.18).
static __device__ __forceinline__ void
async_ld16(unsigned lds_addr, unsigned long long gaddr) {
    asm volatile("global_load_async_to_lds_b128 %0, %1, off"
                 :: "v"(lds_addr), "v"(gaddr) : "memory");
}
static __device__ __forceinline__ void wait_async0() {
    asm volatile("s_wait_asynccnt 0x0" ::: "memory");
}

// Assemble a 16x32 WMMA fragment from an LDS (or any) bf16 tile row pointer:
// fragment elems = p[0..7] and p[16..23]  (two b128 reads).
static __device__ __forceinline__ bf16x16 frag_from_ptr(const __bf16* p) {
    bf16x8 lo = *(const bf16x8*)p;
    bf16x8 hv = *(const bf16x8*)(p + 16);
    return __builtin_shufflevector(lo, hv, 0, 1, 2, 3, 4, 5, 6, 7,
                                           8, 9, 10, 11, 12, 13, 14, 15);
}

// Global fragment load (used by flash attention; per ISA §7.12.2 layout):
//   lane L<16 : row=row0+L,    k = k0 + {0..7, 16..23}
//   lane L>=16: row=row0+L-16, k = k0 + {8..15, 24..31}
static __device__ __forceinline__ bf16x16
load_frag_bf16(const __bf16* __restrict__ base, int ld, int row0, int k0) {
    int lane = threadIdx.x & 31;
    return frag_from_ptr(base + (size_t)(row0 + (lane & 15)) * ld
                              + k0 + ((lane >> 4) << 3));
}

// ---------------------------------------------------------------------------
// Pass 0: bulk f32 -> bf16 (hardware cvt, 8 elem/thread, b128 in / b128 out)
// ---------------------------------------------------------------------------
__global__ __launch_bounds__(256)
void cvt_f32_bf16_kernel(const float* __restrict__ src, __bf16* __restrict__ dst,
                         long n) {
    long i = ((long)blockIdx.x * blockDim.x + threadIdx.x) * 8;
    if (i + 8 <= n) {
        f32x8 v = *(const f32x8*)(src + i);
        bf16x8 o;
#pragma unroll
        for (int e = 0; e < 8; ++e) o[e] = (__bf16)v[e];
        *(bf16x8*)(dst + i) = o;
    }
}

// ---------------------------------------------------------------------------
// Shared GEMM core: block (256 thr / 8 waves) computes C[128 x 64] of
// A[M,K] @ B[N,K]^T, K = WE = 1024, via double-buffered async-LDS tiles.
//   A tile: 128x32 bf16 = 8KB, B tile: 64x32 bf16 = 4KB, x2 buffers = 24KB.
// Each wave owns a 16x64 slice -> acc[4] of f32x8.
// ---------------------------------------------------------------------------
#define GEMM_SHARED_DECL                                                     \
    __shared__ __align__(16) __bf16 Ab[2][128 * 32];                         \
    __shared__ __align__(16) __bf16 Bb[2][64 * 32];

static __device__ __forceinline__ void
gemm_issue(const __bf16* __restrict__ A, const __bf16* __restrict__ Bm,
           __bf16 (*Ab)[128 * 32], __bf16 (*Bb)[64 * 32],
           int m0, int n0, int k0, int buf) {
    int t = threadIdx.x;
    // A: 8KB tile, 32B per thread (two contiguous b128)
    int arow = t >> 1;              // 0..127
    int acol = (t & 1) * 16;        // element col 0 or 16
    unsigned al = (unsigned)(size_t)&Ab[buf][arow * 32 + acol];
    unsigned long long ag =
        (unsigned long long)(size_t)(A + (size_t)(m0 + arow) * WE + k0 + acol);
    async_ld16(al, ag);
    async_ld16(al + 16u, ag + 16ull);
    // B: 4KB tile, 16B per thread
    int brow = t >> 2;              // 0..63
    int bcol = (t & 3) * 8;         // element col 0,8,16,24
    unsigned bl = (unsigned)(size_t)&Bb[buf][brow * 32 + bcol];
    unsigned long long bg =
        (unsigned long long)(size_t)(Bm + (size_t)(n0 + brow) * WE + k0 + bcol);
    async_ld16(bl, bg);
}

static __device__ __forceinline__ void
gemm_compute(const __bf16* __restrict__ Abuf, const __bf16* __restrict__ Bbuf,
             f32x8 acc[4]) {
    int lane = threadIdx.x & 31;
    int wave = threadIdx.x >> 5;
    int r = lane & 15, hk = (lane >> 4) << 3;
    bf16x16 af = frag_from_ptr(Abuf + (wave * 16 + r) * 32 + hk);
#pragma unroll
    for (int t4 = 0; t4 < 4; ++t4) {
        bf16x16 bfr = frag_from_ptr(Bbuf + (t4 * 16 + r) * 32 + hk);
        acc[t4] = wmma_bf16(af, bfr, acc[t4]);
    }
}

#define GEMM_MAIN_LOOP(Aptr, Bptr, m0, n0)                                   \
    f32x8 acc[4] = {f32x8{}, f32x8{}, f32x8{}, f32x8{}};                     \
    gemm_issue(Aptr, Bptr, Ab, Bb, m0, n0, 0, 0);                            \
    for (int kt = 0; kt < WE / 32; ++kt) {                                   \
        int buf = kt & 1;                                                    \
        wait_async0();                                                       \
        __syncthreads();          /* tile[buf] visible to all waves */       \
        if ((kt + 1) * 32 < WE)                                              \
            gemm_issue(Aptr, Bptr, Ab, Bb, m0, n0, (kt + 1) * 32, buf ^ 1);  \
        gemm_compute(Ab[buf], Bb[buf], acc);                                 \
        __syncthreads();          /* done reading before it is overwritten */\
    }

// ---------------------------------------------------------------------------
// Pass 1: qkv projection (M=8192, N=3072, K=1024).
// Epilogue scatters bf16 q,k in (B,H,S,D) and vT in (B,H,D,S).
// ---------------------------------------------------------------------------
__global__ __launch_bounds__(256)
void qkv_proj_kernel(const __bf16* __restrict__ x, const __bf16* __restrict__ W,
                     __bf16* __restrict__ qbuf, __bf16* __restrict__ kbuf,
                     __bf16* __restrict__ vtbuf) {
    GEMM_SHARED_DECL
    int m0 = blockIdx.x * 128;
    int n0 = blockIdx.y * 64;
    GEMM_MAIN_LOOP(x, W, m0, n0)

    int lane = threadIdx.x & 31;
    int wave = threadIdx.x >> 5;
    int cn = lane & 15, hi = lane >> 4;
    int mw = m0 + wave * 16;
    int b = mw / WS;
    int sbase = mw % WS;
#pragma unroll
    for (int t4 = 0; t4 < 4; ++t4) {
        int n = n0 + t4 * 16 + cn;
        int which = n >> 10;          // 0=q 1=k 2=v (uniform per tile)
        int e = n & (WE - 1);
        int hh = e >> 6;
        int d  = e & (WD - 1);
#pragma unroll
        for (int i = 0; i < 8; ++i) {
            int s = sbase + i + 8 * hi;
            __bf16 v = (__bf16)acc[t4][i];
            if (which == 0)
                qbuf[(((size_t)b * WH + hh) * WS + s) * WD + d] = v;
            else if (which == 1)
                kbuf[(((size_t)b * WH + hh) * WS + s) * WD + d] = v;
            else
                vtbuf[(((size_t)b * WH + hh) * WD + d) * WS + s] = v;
        }
    }
}

// ---------------------------------------------------------------------------
// Pass 3 kernel: out = attn_out @ Wout^T (M=8192, N=1024), f32 output.
// ---------------------------------------------------------------------------
__global__ __launch_bounds__(256)
void out_proj_kernel(const __bf16* __restrict__ ao, const __bf16* __restrict__ W,
                     float* __restrict__ out) {
    GEMM_SHARED_DECL
    int m0 = blockIdx.x * 128;
    int n0 = blockIdx.y * 64;
    GEMM_MAIN_LOOP(ao, W, m0, n0)

    int lane = threadIdx.x & 31;
    int wave = threadIdx.x >> 5;
    int cn = lane & 15, hi = lane >> 4;
    int mw = m0 + wave * 16;
#pragma unroll
    for (int t4 = 0; t4 < 4; ++t4)
#pragma unroll
        for (int i = 0; i < 8; ++i)
            out[(size_t)(mw + i + 8 * hi) * WE + n0 + t4 * 16 + cn] = acc[t4][i];
}

// ---------------------------------------------------------------------------
// Pass 2: flash attention, one wave per (b,h, 16-query tile).
// ---------------------------------------------------------------------------
__global__ __launch_bounds__(32)
void flash_attn_kernel(const __bf16* __restrict__ qbuf,
                       const __bf16* __restrict__ kbuf,
                       const __bf16* __restrict__ vtbuf,
                       const int* __restrict__ seq_lengths,
                       const int* __restrict__ wlp, const int* __restrict__ wrp,
                       __bf16* __restrict__ aobuf) {
    __shared__ __align__(16) __bf16 pstage[16 * 32];   // probs, C->A transpose

    const int tiles = WS / 16;
    int tile = blockIdx.x;
    int i0 = (tile % tiles) * 16;
    int bh = tile / tiles;
    int h = bh % WH, b = bh / WH;
    int sl = seq_lengths[b];
    int wl = wlp[0], wr = wrp[0];

    int lane = threadIdx.x & 31;
    int cn = lane & 15, hi = lane >> 4;

    const __bf16* qb  = qbuf  + (size_t)bh * WS * WD;
    const __bf16* kb  = kbuf  + (size_t)bh * WS * WD;
    const __bf16* vtb = vtbuf + (size_t)bh * WD * WS;

    bf16x16 qf0 = load_frag_bf16(qb, WD, i0, 0);
    bf16x16 qf1 = load_frag_bf16(qb, WD, i0, 32);

    f32x8 o0 = {}, o1 = {}, o2 = {}, o3 = {};
    float mrow[8], lrow[8];
#pragma unroll
    for (int i = 0; i < 8; ++i) { mrow[i] = -3.0e38f; lrow[i] = 0.0f; }

    int jstart = 0;
    if (wl >= 0) { jstart = i0 - wl; if (jstart < 0) jstart = 0; }
    int jend = sl - 1;
    int jcap = (wr >= 0) ? (i0 + 15 + wr) : (WS - 1);
    if (jend > jcap) jend = jcap;
    if (jend > WS - 1) jend = WS - 1;
    jstart &= ~31;

    const float scale = 0.125f;   // 1/sqrt(D=64)

    for (int j0 = jstart; j0 <= jend; j0 += 32) {
        // ---- scores S[16q][32k], contraction over D=64 ----
        f32x8 s_lo = {}, s_hi = {};
        s_lo = wmma_bf16(qf0, load_frag_bf16(kb, WD, j0,      0),  s_lo);
        s_lo = wmma_bf16(qf1, load_frag_bf16(kb, WD, j0,      32), s_lo);
        s_hi = wmma_bf16(qf0, load_frag_bf16(kb, WD, j0 + 16, 0),  s_hi);
        s_hi = wmma_bf16(qf1, load_frag_bf16(kb, WD, j0 + 16, 32), s_hi);

        // ---- scale + window/padding mask (branch-free) ----
        float sv[2][8];
#pragma unroll
        for (int t = 0; t < 2; ++t) {
            int j = j0 + t * 16 + cn;
#pragma unroll
            for (int i = 0; i < 8; ++i) {
                int m = i0 + i + 8 * hi;
                float v = (t ? s_hi[i] : s_lo[i]) * scale;
                bool ok = (j < sl);
                if (wl >= 0) ok = ok && (j >= m - wl);
                if (wr >= 0) ok = ok && (j <= m + wr);
                sv[t][i] = ok ? v : -1.0e30f;
            }
        }

        // ---- online softmax (wave32 shfl_xor row reductions) ----
        float a[8], mnew[8];
#pragma unroll
        for (int i = 0; i < 8; ++i) {
            float mx = fmaxf(sv[0][i], sv[1][i]);
            mx = fmaxf(mx, __shfl_xor(mx, 1));
            mx = fmaxf(mx, __shfl_xor(mx, 2));
            mx = fmaxf(mx, __shfl_xor(mx, 4));
            mx = fmaxf(mx, __shfl_xor(mx, 8));
            mnew[i] = fmaxf(mrow[i], mx);
            a[i] = __expf(mrow[i] - mnew[i]);
            mrow[i] = mnew[i];
        }
#pragma unroll
        for (int i = 0; i < 8; ++i) {
            float p0 = __expf(sv[0][i] - mnew[i]);
            float p1 = __expf(sv[1][i] - mnew[i]);
            float rs = p0 + p1;
            rs += __shfl_xor(rs, 1);
            rs += __shfl_xor(rs, 2);
            rs += __shfl_xor(rs, 4);
            rs += __shfl_xor(rs, 8);
            lrow[i] = lrow[i] * a[i] + rs;
            int r = i + 8 * hi;                    // C-layout row
            pstage[r * 32 + cn]      = (__bf16)p0;
            pstage[r * 32 + 16 + cn] = (__bf16)p1;
        }
        __syncthreads();

        // ---- rescale O accumulators ----
#pragma unroll
        for (int i = 0; i < 8; ++i) {
            o0[i] *= a[i]; o1[i] *= a[i]; o2[i] *= a[i]; o3[i] *= a[i];
        }

        // ---- reload P in A-fragment layout (two ds_load_b128) ----
        bf16x16 pf = frag_from_ptr(&pstage[(lane & 15) * 32 + (hi << 3)]);

        // ---- O += P(16x32) @ V(32x64); V transposed => contiguous keys ----
        o0 = wmma_bf16(pf, load_frag_bf16(vtb, WS,  0, j0), o0);
        o1 = wmma_bf16(pf, load_frag_bf16(vtb, WS, 16, j0), o1);
        o2 = wmma_bf16(pf, load_frag_bf16(vtb, WS, 32, j0), o2);
        o3 = wmma_bf16(pf, load_frag_bf16(vtb, WS, 48, j0), o3);
        __syncthreads();   // protect pstage before next block overwrites it
    }

    // ---- normalize, write attn_out (B,S,E) bf16 ----
#pragma unroll
    for (int i = 0; i < 8; ++i) {
        float inv = (lrow[i] > 0.0f) ? (1.0f / lrow[i]) : 0.0f;
        int s = i0 + i + 8 * hi;
        size_t rb = ((size_t)b * WS + s) * WE + h * WD;
        aobuf[rb +  0 + cn] = (__bf16)(o0[i] * inv);
        aobuf[rb + 16 + cn] = (__bf16)(o1[i] * inv);
        aobuf[rb + 32 + cn] = (__bf16)(o2[i] * inv);
        aobuf[rb + 48 + cn] = (__bf16)(o3[i] * inv);
    }
}

// ---------------------------------------------------------------------------
extern "C" void kernel_launch(void* const* d_in, const int* in_sizes, int n_in,
                              void* d_out, int out_size, void* d_ws, size_t ws_size,
                              hipStream_t stream) {
    const float* x    = (const float*)d_in[0];   // (B,S,E)
    const float* Wqkv = (const float*)d_in[1];   // (3E,E)
    const float* Wout = (const float*)d_in[2];   // (E,E)
    const int*   seq  = (const int*)d_in[3];     // (B,)
    const int*   wl   = (const int*)d_in[4];     // scalar
    const int*   wr   = (const int*)d_in[5];     // scalar
    float* out = (float*)d_out;
    (void)in_sizes; (void)n_in; (void)out_size; (void)ws_size;

    // workspace carve-up (all bf16, all 16B-aligned)
    const size_t nx  = (size_t)WB * WS * WE;     // 8388608
    const size_t nw1 = (size_t)WN3 * WE;         // 3145728
    const size_t nw2 = (size_t)WE * WE;          // 1048576
    __bf16* xb   = (__bf16*)d_ws;
    __bf16* w1b  = xb + nx;
    __bf16* w2b  = w1b + nw1;
    __bf16* qb   = w2b + nw2;
    __bf16* kb   = qb + nx;
    __bf16* vtb  = kb + nx;
    __bf16* aob  = vtb + nx;

    // Pass 0: one-time f32 -> bf16 conversions
    cvt_f32_bf16_kernel<<<dim3((unsigned)(nx  / 8 / 256)), dim3(256), 0, stream>>>(x,    xb,  (long)nx);
    cvt_f32_bf16_kernel<<<dim3((unsigned)(nw1 / 8 / 256)), dim3(256), 0, stream>>>(Wqkv, w1b, (long)nw1);
    cvt_f32_bf16_kernel<<<dim3((unsigned)(nw2 / 8 / 256)), dim3(256), 0, stream>>>(Wout, w2b, (long)nw2);

    // Pass 1: QKV projection: M=8192, N=3072, K=1024
    qkv_proj_kernel<<<dim3((WB * WS) / 128, WN3 / 64), dim3(256), 0, stream>>>(
        xb, w1b, qb, kb, vtb);

    // Pass 2: flash attention: one wave per (b,h,16-query tile)
    flash_attn_kernel<<<dim3(WB * WH * (WS / 16)), dim3(32), 0, stream>>>(
        qb, kb, vtb, seq, wl, wr, aob);

    // Pass 3: output projection: M=8192, N=1024, K=1024
    out_proj_kernel<<<dim3((WB * WS) / 128, WE / 64), dim3(256), 0, stream>>>(
        aob, w2b, out);
}
